// RNN_20151986553140
// MI455X (gfx1250) — matrix operand order repaired and checked by reading, not compiled, not run on hardware
//
#include <hip/hip_runtime.h>

// ---------------------------------------------------------------------------
// 2-layer ReLU RNN on MI455X (gfx1250), persistent-kernel design.
//   h0 = relu(x_t @ Wih0^T + b0 + h0 @ Whh0^T)
//   h1 = relu(h0  @ Wih1^T + b1 + h1 @ Whh1^T)   -> out[b,t,:]
// Phase A (parallel, bandwidth-bound): convert x to bf16 once (64MB->32MB).
// Phase B (persistent, latency-bound): 32 workgroups; WG g owns output
// columns [32g, 32g+32). All four weight slices live in LDS as pre-swizzled
// bf16 WMMA B-fragments (224 KB/WGP of the 320 KB budget). Recurrent state
// ping-pongs in a tiny L2-resident bf16 buffer; one release/acquire grid
// barrier per timestep. The critical loop is pure b128 loads + v_wmma, with
// FOUR independent accumulator chains to hide WMMA latency.
// ---------------------------------------------------------------------------

namespace {
constexpr int kB   = 64;    // batch
constexpr int kT   = 512;   // timesteps
constexpr int kI   = 512;   // input features
constexpr int kH   = 1024;  // hidden
constexpr int kNWG = 32;    // workgroups; each owns kH/kNWG = 32 columns
constexpr int kKB_I = kI / 32;   // 16 K-blocks over input
constexpr int kKB_H = kH / 32;   // 32 K-blocks over hidden

// LDS layout (uint units): swizzled bf16 weight slices (B-matrix fragments).
constexpr int kLdsIH0 = 0;
constexpr int kLdsHH0 = kLdsIH0 + 2 * kKB_I * 32 * 8;  //  8192
constexpr int kLdsIH1 = kLdsHH0 + 2 * kKB_H * 32 * 8;  // 24576
constexpr int kLdsHH1 = kLdsIH1 + 2 * kKB_H * 32 * 8;  // 40960
constexpr int kLdsTot = kLdsHH1 + 2 * kKB_H * 32 * 8;  // 57344 uints = 224 KB
}  // namespace

typedef __attribute__((ext_vector_type(16))) __bf16 v16bf;
typedef __attribute__((ext_vector_type(8)))  float  v8f;

union Frag { unsigned u[8]; v16bf v; };

__device__ __forceinline__ unsigned bf16_rne(float f) {
  unsigned u = __builtin_bit_cast(unsigned, f);
  return (u + 0x7fffu + ((u >> 16) & 1u)) >> 16;  // round-to-nearest-even
}
__device__ __forceinline__ unsigned pack2(float lo, float hi) {
  return bf16_rne(lo) | (bf16_rne(hi) << 16);
}

__device__ __forceinline__ v8f wmma_bf16(const Frag& a, const Frag& b, v8f c) {
  // v_wmma_f32_16x16x32_bf16: D = A(16x32) * B(32x16) + C(16x16 f32)
  return __builtin_amdgcn_wmma_f32_16x16x32_bf16(
      /*neg_a=*/false, a.v, /*neg_b=*/false, b.v,
      /*c_mod=*/(short)0, c, /*reuse_a=*/false, /*reuse_b=*/false);
}

__device__ __forceinline__ void load_a(Frag& a, const unsigned* __restrict__ base,
                                       const int* koffA, int kb) {
#pragma unroll
  for (int j = 0; j < 8; ++j) a.u[j] = base[(kb * 32 + koffA[j]) >> 1];
}
__device__ __forceinline__ void load_b(Frag& b, const unsigned* bp) {
#pragma unroll
  for (int j = 0; j < 8; ++j) b.u[j] = bp[j];
}

// Load W[n0+ntile*16+row, :] slice into LDS as swizzled bf16 B-fragments.
// B fragment order used here (contiguous k; A matches): col = lane%16, lanes 0-15 hold
// K=0..15 (VGPR j -> K=2j,2j+1), lanes 16-31 hold K=16..31.
__device__ void fill_w(unsigned* dst, const float* __restrict__ W,
                       int n0, int K, int wave, int lane) {
  const int nkb  = K >> 5;
  const int half = lane >> 4;
  const int nloc = lane & 15;
  for (int blk = wave; blk < 2 * nkb; blk += 8) {   // 8 waves cooperate
    const int ntile = blk / nkb;
    const int kblk  = blk % nkb;
    const float* wr = W + (size_t)(n0 + ntile * 16 + nloc) * K
                        + kblk * 32 + (half ? 16 : 0);
    unsigned* d = dst + ((size_t)blk * 32 + lane) * 8;
#pragma unroll
    for (int j = 0; j < 8; ++j) d[j] = pack2(wr[2 * j], wr[2 * j + 1]);
  }
}

// Zero recurrent state ping-pong buffers + barrier counter.
__global__ void rnn_init(unsigned* __restrict__ zbase, int* __restrict__ ctr) {
  const unsigned i = blockIdx.x * blockDim.x + threadIdx.x;
  if (i < 2u * (2u * kB * kH / 2u)) zbase[i] = 0u;  // h0[2] + h1[2] bf16
  if (i == 0) *ctr = 0;
}

// Parallel fp32 -> bf16 conversion of x (row-major preserved). 8 floats/thread.
__global__ __launch_bounds__(256) void x_to_bf16(
    const float* __restrict__ x, unsigned* __restrict__ xb) {
  const size_t i = (size_t)blockIdx.x * blockDim.x + threadIdx.x;  // uint4 idx
  const float4* xv = (const float4*)x;
  const float4 a = xv[2 * i];
  const float4 b = xv[2 * i + 1];
  uint4 o;
  o.x = pack2(a.x, a.y);
  o.y = pack2(a.z, a.w);
  o.z = pack2(b.x, b.y);
  o.w = pack2(b.z, b.w);
  ((uint4*)xb)[i] = o;
}

__global__ __launch_bounds__(256, 1) void rnn_persistent(
    const float* __restrict__ b0,     // [H]
    const float* __restrict__ w_ih0,  // [H,I]
    const float* __restrict__ w_hh0,  // [H,H]
    const float* __restrict__ w_ih1,  // [H,H]
    const float* __restrict__ w_hh1,  // [H,H]
    const float* __restrict__ b1,     // [H]
    const unsigned* xb,               // [B,T,I] bf16 (uint pairs)
    float* __restrict__ out,          // [B,T,H]
    unsigned short* h0buf,            // [2][B][H] bf16 ping-pong
    unsigned short* h1buf,            // [2][B][H] bf16 ping-pong
    int* ctr) {
  __shared__ unsigned lds[kLdsTot];

  const int lane  = threadIdx.x & 31;
  const int wave  = threadIdx.x >> 5;   // 8 waves: (mtile 0..3) x (ntile 0..1)
  const int half  = lane >> 4;
  const int mtile = wave & 3;
  const int ntile = wave >> 2;
  const int n0    = blockIdx.x * (kH / kNWG);

  // One-time: stage all four weight slices into LDS (bf16, B-swizzled).
  fill_w(lds + kLdsIH0, w_ih0, n0, kI, wave, lane);
  fill_w(lds + kLdsHH0, w_hh0, n0, kH, wave, lane);
  fill_w(lds + kLdsIH1, w_ih1, n0, kH, wave, lane);
  fill_w(lds + kLdsHH1, w_hh1, n0, kH, wave, lane);
  __syncthreads();

  // A fragment K offsets (16-bit A 16x32): VGPR j -> K=2j+16*half (matches
  // the B fragment order); each uint holds (K, K+1). The eight uints are
  // contiguous half-words -> compiler merges to global_load_b128.
  int koffA[8];
#pragma unroll
  for (int j = 0; j < 8; ++j)
    koffA[j] = 2 * j + 16 * half;

  const int m = mtile * 16 + (lane & 15);        // A row = batch index
  const int n = n0 + ntile * 16 + (lane & 15);   // D column = hidden unit
  const float bias0 = b0[n];
  const float bias1 = b1[n];

  const unsigned* h0u = (const unsigned*)h0buf;  // uint view: [2][B][H/2]
  const unsigned* h1u = (const unsigned*)h1buf;
  const unsigned* xrow = xb + (size_t)m * kT * (kI / 2);  // x[m, :, :] bf16

  const unsigned* Bih0 = lds + kLdsIH0 + ntile * kKB_I * 32 * 8;
  const unsigned* Bhh0 = lds + kLdsHH0 + ntile * kKB_H * 32 * 8;
  const unsigned* Bih1 = lds + kLdsIH1 + ntile * kKB_H * 32 * 8;
  const unsigned* Bhh1 = lds + kLdsHH1 + ntile * kKB_H * 32 * 8;

  for (int t = 0; t < kT; ++t) {
    const int p = t & 1;

    // ---------------- layer 0: relu(x_t@Wih0^T + b0 + h0@Whh0^T) ----------
    // Four independent accumulator chains; bias folded into chain 0's C.
    v8f acc[4];
#pragma unroll
    for (int r = 0; r < 8; ++r) {
      acc[0][r] = bias0;
      acc[1][r] = 0.f;
      acc[2][r] = 0.f;
      acc[3][r] = 0.f;
    }

    const unsigned* xt = xrow + (size_t)t * (kI / 2);
    if (t + 1 < kT) __builtin_prefetch(xt + kI / 2, 0, 0);  // next step's row
    for (int kb = 0; kb < kKB_I; kb += 4) {
#pragma unroll
      for (int q = 0; q < 4; ++q) {
        Frag a, b;
        load_a(a, xt, koffA, kb + q);
        load_b(b, Bih0 + ((kb + q) * 32 + lane) * 8);
        acc[q] = wmma_bf16(a, b, acc[q]);
      }
    }
    const unsigned* h0prev = h0u + p * (kB * kH / 2) + m * (kH / 2);
    for (int kb = 0; kb < kKB_H; kb += 4) {
#pragma unroll
      for (int q = 0; q < 4; ++q) {
        Frag a, b;
        load_a(a, h0prev, koffA, kb + q);
        load_b(b, Bhh0 + ((kb + q) * 32 + lane) * 8);
        acc[q] = wmma_bf16(a, b, acc[q]);
      }
    }
    unsigned short* h0nxt = h0buf + (1 - p) * (kB * kH);
#pragma unroll
    for (int r = 0; r < 8; ++r) {
      const float v =
          fmaxf((acc[0][r] + acc[1][r]) + (acc[2][r] + acc[3][r]), 0.f);
      const int mr = mtile * 16 + r + 8 * half;   // D row layout
      h0nxt[mr * kH + n] = (unsigned short)bf16_rne(v);
    }

    // --------------- grid barrier (release/acquire, monotonic ctr) --------
    __threadfence();
    __syncthreads();
    if (threadIdx.x == 0) {
      atomicAdd(ctr, 1);
      const int target = kNWG * (t + 1);
      while (*(volatile int*)ctr < target) __builtin_amdgcn_s_sleep(1);
    }
    __syncthreads();
    __threadfence();

    // ---------------- layer 1: relu(h0@Wih1^T + b1 + h1@Whh1^T) -----------
    v8f acc1[4];
#pragma unroll
    for (int r = 0; r < 8; ++r) {
      acc1[0][r] = bias1;
      acc1[1][r] = 0.f;
      acc1[2][r] = 0.f;
      acc1[3][r] = 0.f;
    }

    const unsigned* h0new  = h0u + (1 - p) * (kB * kH / 2) + m * (kH / 2);
    const unsigned* h1prev = h1u + p * (kB * kH / 2) + m * (kH / 2);
    for (int kb = 0; kb < kKB_H; kb += 4) {
#pragma unroll
      for (int q = 0; q < 4; ++q) {
        Frag a, b;
        load_a(a, h0new, koffA, kb + q);
        load_b(b, Bih1 + ((kb + q) * 32 + lane) * 8);
        acc1[q] = wmma_bf16(a, b, acc1[q]);
      }
    }
    for (int kb = 0; kb < kKB_H; kb += 4) {
#pragma unroll
      for (int q = 0; q < 4; ++q) {
        Frag a, b;
        load_a(a, h1prev, koffA, kb + q);
        load_b(b, Bhh1 + ((kb + q) * 32 + lane) * 8);
        acc1[q] = wmma_bf16(a, b, acc1[q]);
      }
    }
    unsigned short* h1nxt = h1buf + (1 - p) * (kB * kH);
#pragma unroll
    for (int r = 0; r < 8; ++r) {
      const float v =
          fmaxf((acc1[0][r] + acc1[1][r]) + (acc1[2][r] + acc1[3][r]), 0.f);
      const int mr = mtile * 16 + r + 8 * half;
      h1nxt[mr * kH + n] = (unsigned short)bf16_rne(v);
      out[((size_t)mr * kT + t) * kH + n] = v;   // fp32 output [B,T,H]
    }
  }
}

extern "C" void kernel_launch(void* const* d_in, const int* in_sizes, int n_in,
                              void* d_out, int out_size, void* d_ws,
                              size_t ws_size, hipStream_t stream) {
  const float* x     = (const float*)d_in[0];
  const float* w_ih0 = (const float*)d_in[1];
  const float* w_hh0 = (const float*)d_in[2];
  const float* b0    = (const float*)d_in[3];
  const float* w_ih1 = (const float*)d_in[4];
  const float* w_hh1 = (const float*)d_in[5];
  const float* b1    = (const float*)d_in[6];
  float* out = (float*)d_out;

  // Workspace: [ctr pad 256B][h0: 2*B*H bf16][h1: 2*B*H bf16][x bf16 32MB]
  char* ws = (char*)d_ws;
  int* ctr = (int*)ws;
  unsigned short* h0buf = (unsigned short*)(ws + 256);
  unsigned short* h1buf = (unsigned short*)(ws + 256 + 2 * kB * kH * 2);
  unsigned* xb = (unsigned*)(ws + 256 + 4 * kB * kH * 2);

  // Phase A (parallel): zero state + convert x to bf16.
  rnn_init<<<512, 256, 0, stream>>>((unsigned*)h0buf, ctr);
  // B*T*I floats = 16,777,216 floats -> 2,097,152 uint4 stores; 8192 blocks.
  x_to_bf16<<<(kB * kT * kI / 8) / 256, 256, 0, stream>>>(x, xb);

  // Phase B (persistent recurrence).
  rnn_persistent<<<kNWG, 256, 0, stream>>>(b0, w_ih0, w_hh0, w_ih1, w_hh1, b1,
                                           xb, out, h0buf, h1buf, ctr);
}
